// Model_19198503813410
// MI455X (gfx1250) — matrix operand order repaired and checked by reading, not compiled, not run on hardware
//
#include <hip/hip_runtime.h>
#include <hip/hip_bf16.h>

// ---------- types ----------
typedef __attribute__((ext_vector_type(16))) __bf16          v16bf;
typedef __attribute__((ext_vector_type(8)))  float           floatx8;   // v8f accumulator
typedef __attribute__((ext_vector_type(16))) unsigned short  ushort16;
typedef __attribute__((ext_vector_type(8)))  unsigned short  ushort8;

#define B_DIM 256
#define T_DIM 256
#define I_DIM 1024
#define H_DIM 1024

__device__ __forceinline__ unsigned short f2bfbits(float f) {
    __bf16 h = (__bf16)f;                       // hardware RNE convert on gfx1250
    return __builtin_bit_cast(unsigned short, h);
}

// ---------- prep: convert weights/hidden to bf16, fuse biases ----------
__global__ void prep_kernel(const float* __restrict__ Wih, const float* __restrict__ Whh,
                            const float* __restrict__ bih, const float* __restrict__ bhh,
                            const float* __restrict__ hidden,
                            unsigned short* __restrict__ WihB, unsigned short* __restrict__ WhhB,
                            float* __restrict__ bias, unsigned short* __restrict__ h0B) {
    int i = blockIdx.x * blockDim.x + threadIdx.x;
    int stride = gridDim.x * blockDim.x;
    for (int k = i; k < H_DIM * I_DIM; k += stride) {
        WihB[k] = f2bfbits(Wih[k]);
        WhhB[k] = f2bfbits(Whh[k]);
    }
    for (int k = i; k < B_DIM * H_DIM; k += stride) h0B[k] = f2bfbits(hidden[k]);
    for (int k = i; k < H_DIM; k += stride) bias[k] = bih[k] + bhh[k];
}

// ---------- one recurrence step ----------
// h_new = tanh( x_t @ W_ih^T + h @ W_hh^T + (b_ih + b_hh) )
// Two waves per 16x16 tile: part 0 does the x-GEMM (K=1024, + bias),
// part 1 does the h-GEMM (K=1024) and passes its partial through LDS.
// 256 blocks x 8 waves = 2048 waves/step -> 2 waves per SIMD32 for latency hiding.
__global__ __launch_bounds__(256) void rnn_step_kernel(
        const float*          __restrict__ x,        // (B, T, I) fp32
        int                   t,
        const unsigned short* __restrict__ hinB,     // (B, H) bf16 bits
        const unsigned short* __restrict__ WihB,     // (H, I) bf16 bits, row-major (K contiguous)
        const unsigned short* __restrict__ WhhB,     // (H, H) bf16 bits
        const float*          __restrict__ bias,     // (H)
        unsigned short*       __restrict__ houtB,    // (B, H) bf16 bits
        float*                __restrict__ houtF,    // (B, H) fp32
        float*                __restrict__ lastbuf)  // (T, H) fp32 : batch row 255 per step
{
    __shared__ float partial[4][32][8];              // [tileInBlock][lane][acc] = 4 KB

    const int wave  = threadIdx.x >> 5;              // 0..7
    const int lane  = threadIdx.x & 31;
    const int part  = wave >> 2;                     // 0: x-GEMM, 1: h-GEMM
    const int tib   = wave & 3;                      // tile within block
    const int tile  = blockIdx.x * 4 + tib;          // 0..1023
    const int m0    = (tile & 15) << 4;              // batch tile base
    const int n0    = (tile >> 4) << 4;              // hidden-unit tile base
    const int row   = lane & 15;
    const int half  = lane >> 4;

    floatx8 c;

    if (part == 0) {
        // -------- input GEMM: K over I, convert fp32 x -> bf16 inline --------
        float bv = bias[n0 + row];
        #pragma unroll
        for (int r = 0; r < 8; ++r) c[r] = bv;

        const float*          xrow = x    + ((size_t)(m0 + row) * T_DIM + t) * I_DIM;
        const unsigned short* wihr = WihB + (size_t)(n0 + row) * I_DIM;

        for (int kb = 0; kb < I_DIM; kb += 32) {
            const int ka = kb + half * 16;
            floatx8 lo = *(const floatx8*)(xrow + ka);        // K = ka .. ka+7
            floatx8 hi = *(const floatx8*)(xrow + ka + 8);    // K = ka+8 .. ka+15
            v16bf a;
            #pragma unroll
            for (int r = 0; r < 8; ++r) { a[r] = (__bf16)lo[r]; a[8 + r] = (__bf16)hi[r]; }
            v16bf b = __builtin_bit_cast(v16bf, *(const ushort16*)(wihr + kb + half * 16));
            c = __builtin_amdgcn_wmma_f32_16x16x32_bf16(false, a, false, b, (short)0, c,
                                                        false, false);
        }
    } else {
        // -------- recurrent GEMM: K over H, h already bf16 --------
        #pragma unroll
        for (int r = 0; r < 8; ++r) c[r] = 0.f;

        const unsigned short* hrow = hinB + (size_t)(m0 + row) * H_DIM;
        const unsigned short* whhr = WhhB + (size_t)(n0 + row) * H_DIM;

        for (int kb = 0; kb < H_DIM; kb += 32) {
            const int ka = kb + half * 16;
            ushort8 lo = *(const ushort8*)(hrow + ka);
            ushort8 hi = *(const ushort8*)(hrow + ka + 8);
            ushort16 au = __builtin_shufflevector(lo, hi,
                              0,1,2,3,4,5,6,7,8,9,10,11,12,13,14,15);
            v16bf a = __builtin_bit_cast(v16bf, au);
            v16bf b = __builtin_bit_cast(v16bf, *(const ushort16*)(whhr + kb + half * 16));
            c = __builtin_amdgcn_wmma_f32_16x16x32_bf16(false, a, false, b, (short)0, c,
                                                        false, false);
        }
        *(floatx8*)&partial[tib][lane][0] = c;       // hand partial to part-0 wave
    }

    __syncthreads();

    if (part == 0) {
        floatx8 p = *(const floatx8*)&partial[tib][lane][0];
        const int n = n0 + row;
        #pragma unroll
        for (int r = 0; r < 8; ++r) {
            const int m = m0 + r + 8 * half;         // C layout: VGPR r, half -> M
            float v = tanhf(c[r] + p[r]);
            houtF[(size_t)m * H_DIM + n] = v;
            houtB[(size_t)m * H_DIM + n] = f2bfbits(v);
            if (m == B_DIM - 1) lastbuf[(size_t)t * H_DIM + n] = v;
        }
    }
}

// ---------- final projection: out = last @ fc_w^T + fc_b (one block per t) ----------
__global__ __launch_bounds__(256) void finalize_kernel(const float* __restrict__ lastbuf,
                                                       const float* __restrict__ fcw,
                                                       const float* __restrict__ fcb,
                                                       float* __restrict__ out) {
    __shared__ float s0[256], s1[256];
    const int t = blockIdx.x;
    const int j = threadIdx.x;
    const float* l = lastbuf + (size_t)t * H_DIM;
    float a0 = 0.f, a1 = 0.f;
    #pragma unroll
    for (int kk = 0; kk < H_DIM; kk += 256) {
        float v = l[j + kk];
        a0 += v * fcw[j + kk];
        a1 += v * fcw[H_DIM + j + kk];
    }
    s0[j] = a0; s1[j] = a1;
    __syncthreads();
    for (int off = 128; off > 0; off >>= 1) {
        if (j < off) { s0[j] += s0[j + off]; s1[j] += s1[j + off]; }
        __syncthreads();
    }
    if (j == 0) {
        out[t * 2 + 0] = s0[0] + fcb[0];
        out[t * 2 + 1] = s1[0] + fcb[1];
    }
}

extern "C" void kernel_launch(void* const* d_in, const int* in_sizes, int n_in,
                              void* d_out, int out_size, void* d_ws, size_t ws_size,
                              hipStream_t stream) {
    (void)in_sizes; (void)n_in; (void)out_size; (void)ws_size;
    const float* x      = (const float*)d_in[0];  // (256,256,1024)
    const float* hidden = (const float*)d_in[1];  // (256,1024)
    const float* Wih    = (const float*)d_in[2];  // (1024,1024)
    const float* bih    = (const float*)d_in[3];  // (1024)
    const float* Whh    = (const float*)d_in[4];  // (1024,1024)
    const float* bhh    = (const float*)d_in[5];  // (1024)
    const float* fcw    = (const float*)d_in[6];  // (2,1024)
    const float* fcb    = (const float*)d_in[7];  // (2)
    float* out = (float*)d_out;                   // [out(256*2) | h_last(256*1024)]

    char* ws = (char*)d_ws;
    unsigned short* WihB = (unsigned short*)ws;  ws += (size_t)H_DIM * I_DIM * 2;
    unsigned short* WhhB = (unsigned short*)ws;  ws += (size_t)H_DIM * H_DIM * 2;
    float*          bias = (float*)ws;           ws += (size_t)H_DIM * 4;
    unsigned short* hb0  = (unsigned short*)ws;  ws += (size_t)B_DIM * H_DIM * 2;
    unsigned short* hb1  = (unsigned short*)ws;  ws += (size_t)B_DIM * H_DIM * 2;
    float*          hf   = (float*)ws;           ws += (size_t)B_DIM * H_DIM * 4;
    float*          last = (float*)ws;           ws += (size_t)T_DIM * H_DIM * 4;

    prep_kernel<<<512, 256, 0, stream>>>(Wih, Whh, bih, bhh, hidden, WihB, WhhB, bias, hb0);

    for (int t = 0; t < T_DIM; ++t) {
        const unsigned short* hin = (t & 1) ? hb1 : hb0;
        unsigned short*       hob = (t & 1) ? hb0 : hb1;
        // 1024 tiles, 4 tiles/block, 2 waves/tile -> 256 blocks of 256 threads
        rnn_step_kernel<<<256, 256, 0, stream>>>(x, t, hin, WihB, WhhB, bias, hob, hf, last);
    }

    finalize_kernel<<<T_DIM, 256, 0, stream>>>(last, fcw, fcb, out);
    hipMemcpyAsync(out + T_DIM * 2, hf, (size_t)B_DIM * H_DIM * sizeof(float),
                   hipMemcpyDeviceToDevice, stream);
}
